// HformerBlockWaveletInfused_206158430304
// MI455X (gfx1250) — matrix operand order repaired and checked by reading, not compiled, not run on hardware
//
#include <hip/hip_runtime.h>

// ---------------------------------------------------------------------------
// Types for CDNA5 (gfx1250) wave32 WMMA
// ---------------------------------------------------------------------------
typedef __attribute__((ext_vector_type(16))) __bf16 v16bf;
typedef __attribute__((ext_vector_type(8)))  __bf16 v8bf;
typedef __attribute__((ext_vector_type(8)))  float  v8f;

#define B_   8
#define H_   256
#define W_   256
#define C_   64
#define HP_  128
#define WP_  128
#define NP_  (HP_ * WP_)          // 16384 pooled pixels per batch
#define NB_  4096                 // band pixels per batch (64*64)
#define NROWS_FULL (B_ * H_ * W_) // 524288
#define NROWS_MP   (B_ * NP_)     // 131072

// db4 decomposition filters (hi[k] = (-1)^k * lo[7-k])
__constant__ float DEC_LO[8] = {
    -0.010597401784997278f,  0.032883011666982945f,  0.030841381835986965f,
    -0.18703481171888114f,  -0.02798376941698385f,   0.6308807679295904f,
     0.7148465705525415f,    0.23037781330885523f };
__constant__ float DEC_HI[8] = {
     0.23037781330885523f,  -0.7148465705525415f,    0.6308807679295904f,
     0.02798376941698385f,  -0.18703481171888114f,  -0.030841381835986965f,
     0.032883011666982945f,  0.010597401784997278f };

// ---------------------------------------------------------------------------
// Helpers
// ---------------------------------------------------------------------------
static __device__ __forceinline__ __bf16 f2bf(float f) {
    unsigned u = __builtin_bit_cast(unsigned, f);
    unsigned r = u + 0x7FFFu + ((u >> 16) & 1u);   // round-to-nearest-even
    unsigned short h = (unsigned short)(r >> 16);
    return __builtin_bit_cast(__bf16, h);
}

// ---- fragment-packed ("swizzled") layouts --------------------------------
// B operand, 32x16 tile: value for (k,n) lives at [tile][lane][e], lane =
// ((k>>3)&1)*16 + (n&15), e = ((k>>4)<<3) | (k&7).  One 32B contiguous load per lane.
static __device__ __forceinline__ size_t bpack_idx(int k, int n, int Nt /*N/16*/) {
    int kk = k & 31;
    int lane = (((kk >> 3) & 1) << 4) | (n & 15);
    return ((size_t)((k >> 5) * Nt + (n >> 4)) * 32 + lane) * 16 + (((kk >> 4) << 3) | (kk & 7));
}
// A^T operand (A[m][k] = src[k][m]), 16x32 tile: lane = ((k>>3)&1)*16 + (m&15),
// e = ((k>>4)<<3) | (k&7).
static __device__ __forceinline__ size_t apack_idx(int k, int m, int Mt /*M/16*/) {
    int kk = k & 31;
    int e    = ((kk >> 4) << 3) | (kk & 7);
    int lane = (((kk >> 3) & 1) << 4) | (m & 15);
    return ((size_t)((k >> 5) * Mt + (m >> 4)) * 32 + lane) * 16 + e;
}
// one 32-byte vector load per lane (lowered to 2x b128)
static __device__ __forceinline__ v16bf load_packed(const __bf16* p, int lane) {
    return *(const v16bf*)(p + lane * 16);
}
// A fragment from row-major memory: per lane it is two contiguous 16B runs.
static __device__ __forceinline__ v16bf load_a_rm(const __bf16* p, int ld, int lane) {
    int half = (lane >> 4) & 1, m = lane & 15;
    const __bf16* row = p + m * ld + half * 8;
    v8bf lo = *(const v8bf*)(row);
    v8bf hi = *(const v8bf*)(row + 16);
    return __builtin_shufflevector(lo, hi, 0, 1, 2, 3, 4, 5, 6, 7,
                                           8, 9, 10, 11, 12, 13, 14, 15);
}
// D layout: element r of v8f  ->  row (r + 8*half), col (lane&15)
static __device__ __forceinline__ void store_d_f32(float* p, int ld, int lane, v8f d) {
    int half = (lane >> 4) & 1, n = lane & 15;
#pragma unroll
    for (int r = 0; r < 8; ++r) p[(r + 8 * half) * ld + n] = d[r];
}
static __device__ __forceinline__ void store_d_bf16(__bf16* p, int ld, int lane, v8f d) {
    int half = (lane >> 4) & 1, n = lane & 15;
#pragma unroll
    for (int r = 0; r < 8; ++r) p[(r + 8 * half) * ld + n] = f2bf(d[r]);
}
static __device__ __forceinline__ v8f wmma_bf16(v16bf a, v16bf b, v8f c) {
    return __builtin_amdgcn_wmma_f32_16x16x32_bf16(false, a, false, b, (short)0, c,
                                                   false, false);
}

// ---------------------------------------------------------------------------
// K0: fp32 -> bf16 weight conversion into B-packed layout
// ---------------------------------------------------------------------------
__global__ void k_cvt_bpack(const float* __restrict__ src, __bf16* __restrict__ dst,
                            int total, int logN) {
    int t = blockIdx.x * blockDim.x + threadIdx.x;
    if (t >= total) return;
    int n = t & ((1 << logN) - 1);
    int k = t >> logN;
    dst[bpack_idx(k, n, 1 << (logN - 4))] = f2bf(src[t]);
}

// ---------------------------------------------------------------------------
// K1: 7x7 depthwise conv, SAME zero padding   -> pa [B,H,W,C] fp32
// ---------------------------------------------------------------------------
__global__ void k_dwconv(const float* __restrict__ img, const float* __restrict__ ker,
                         float* __restrict__ pa) {
    int t = blockIdx.x * blockDim.x + threadIdx.x;
    int c = t & 63, w = (t >> 6) & 255, h = (t >> 14) & 255, b = t >> 22;
    float acc = 0.f;
#pragma unroll
    for (int r = 0; r < 7; ++r) {
        int hh = h + r - 3;
        if (hh < 0 || hh >= H_) continue;
#pragma unroll
        for (int s = 0; s < 7; ++s) {
            int ww = w + s - 3;
            if (ww < 0 || ww >= W_) continue;
            acc += img[(((b * H_ + hh) * W_) + ww) * C_ + c] * ker[(r * 7 + s) * C_ + c];
        }
    }
    pa[t] = acc;
}

// ---------------------------------------------------------------------------
// K2: 2x2 maxpool -> mp fp32 [B,128,128,C] + mpb bf16 (row-major A operand)
// ---------------------------------------------------------------------------
__global__ void k_maxpool(const float* __restrict__ img, float* __restrict__ mp,
                          __bf16* __restrict__ mpb) {
    int t = blockIdx.x * blockDim.x + threadIdx.x;
    int c = t & 63, j = (t >> 6) & 127, i = (t >> 13) & 127, b = t >> 20;
    const float* p = img + (((b * H_ + 2 * i) * W_) + 2 * j) * C_ + c;
    float m = fmaxf(fmaxf(p[0], p[C_]), fmaxf(p[W_ * C_], p[W_ * C_ + C_]));
    mp[t] = m;
    mpb[t] = f2bf(m);
}

// ---------------------------------------------------------------------------
// K3: periodized db4 DWT along H: mp [B,128,128,C] -> lo,hi [B,64,128,C]
// ---------------------------------------------------------------------------
__global__ void k_dwt_h(const float* __restrict__ mp, float* __restrict__ lo,
                        float* __restrict__ hi) {
    int t = blockIdx.x * blockDim.x + threadIdx.x;
    int c = t & 63, w = (t >> 6) & 127, m = (t >> 13) & 63, b = t >> 19;
    float a = 0.f, d = 0.f;
#pragma unroll
    for (int k = 0; k < 8; ++k) {
        int row = (2 * m + k) & 127;
        float x = mp[(((b * HP_ + row) * WP_) + w) * C_ + c];
        a += DEC_LO[k] * x;
        d += DEC_HI[k] * x;
    }
    lo[t] = a;
    hi[t] = d;
}

// ---------------------------------------------------------------------------
// K4: DWT along W. Writes each band element into BOTH packed operand buffers:
//   bandsA : A^T-packed  (Gram A operand, A[m][k] = fb[k][m])
//   bandsB : B-packed    (Gram B operand)
// Per (band,batch) region is NB_*C_ elements.
// ---------------------------------------------------------------------------
__global__ void k_dwt_w(const float* __restrict__ lo, const float* __restrict__ hi,
                        __bf16* __restrict__ bandsA, __bf16* __restrict__ bandsB) {
    int pair = blockIdx.y;                     // 0: lo -> bands 0,1; 1: hi -> 2,3
    const float* src = pair ? hi : lo;
    int t = blockIdx.x * blockDim.x + threadIdx.x;
    int c = t & 63, j = (t >> 6) & 63, m = (t >> 12) & 63, b = t >> 18;
    float a = 0.f, d = 0.f;
#pragma unroll
    for (int k = 0; k < 8; ++k) {
        int col = (2 * j + k) & 127;
        float x = src[(((b * 64 + m) * WP_) + col) * C_ + c];
        a += DEC_LO[k] * x;
        d += DEC_HI[k] * x;
    }
    int row = m * 64 + j;                      // 0..4095 (K index)
    size_t regA = ((size_t)((pair * 2 + 0) * B_ + b)) * NB_ * C_;
    size_t regD = ((size_t)((pair * 2 + 1) * B_ + b)) * NB_ * C_;
    size_t ia = apack_idx(row, c, 4);
    size_t ib = bpack_idx(row, c, 4);
    __bf16 ab = f2bf(a), db = f2bf(d);
    bandsA[regA + ia] = ab;  bandsB[regA + ib] = ab;
    bandsA[regD + ia] = db;  bandsB[regD + ib] = db;
}

// ---------------------------------------------------------------------------
// K5: Gram matrix G = fb^T fb per (band,batch). WMMA, K = 4096.
// grid: (batch=8, band=4), block 256. Wave w: mt = w>>1, nt in {2*(w&1), +1}.
// 3 packed vector loads per 2 WMMAs.
// ---------------------------------------------------------------------------
__global__ void k_gram(const __bf16* __restrict__ bandsA,
                       const __bf16* __restrict__ bandsB, float* __restrict__ G) {
    int b = blockIdx.x, band = blockIdx.y;
    int wave = threadIdx.x >> 5, lane = threadIdx.x & 31;
    size_t reg = ((size_t)(band * B_ + b)) * NB_ * C_;
    const __bf16* Ap = bandsA + reg;
    const __bf16* Bp = bandsB + reg;
    float* Gp = G + ((size_t)(band * B_ + b)) * C_ * C_;   // [64][64]
    int mt = wave >> 1;
    int nt0 = (wave & 1) * 2;
    v8f acc0 = {}, acc1 = {};
    for (int kt = 0; kt < 128; ++kt) {
        v16bf a  = load_packed(Ap + (size_t)(kt * 4 + mt) * 512, lane);
        v16bf b0 = load_packed(Bp + (size_t)(kt * 4 + nt0) * 512, lane);
        v16bf b1 = load_packed(Bp + (size_t)(kt * 4 + nt0 + 1) * 512, lane);
        acc0 = wmma_bf16(a, b0, acc0);
        acc1 = wmma_bf16(a, b1, acc1);
    }
    store_d_f32(Gp + (mt * 16) * C_ + nt0 * 16, C_, lane, acc0);
    store_d_f32(Gp + (mt * 16) * C_ + (nt0 + 1) * 16, C_, lane, acc1);
}

// ---------------------------------------------------------------------------
// K6: att = softmax( Wq^T G Wk / 8 ) rows, per (band,batch).
// Output written directly in B-packed bf16 layout for the v@att GEMM.
// ---------------------------------------------------------------------------
__global__ void k_att(const float* __restrict__ G, const float* __restrict__ Wq,
                      const float* __restrict__ Wk, __bf16* __restrict__ att) {
    __shared__ float sG[4096];
    __shared__ float sT[4096];
    int b = blockIdx.x, band = blockIdx.y;
    int tid = threadIdx.x;
    const float* Gp  = G  + ((size_t)(band * B_ + b)) * 4096;
    const float* Wqp = Wq + (size_t)band * 4096;   // [a][c]
    const float* Wkp = Wk + (size_t)band * 4096;   // [e][d]
    for (int i = tid; i < 4096; i += 256) sG[i] = Gp[i];
    __syncthreads();
    // T1[c][e] = sum_a Wq[a][c] * G[a][e]
    for (int i = tid * 16; i < tid * 16 + 16; ++i) {
        int c = i >> 6, e = i & 63;
        float s = 0.f;
        for (int a = 0; a < 64; ++a) s += Wqp[a * 64 + c] * sG[a * 64 + e];
        sT[i] = s;
    }
    __syncthreads();
    // att0[c][d] = (sum_e T1[c][e] * Wk[e][d]) / 8  (overwrite sG)
    for (int i = tid * 16; i < tid * 16 + 16; ++i) {
        int c = i >> 6, d = i & 63;
        float s = 0.f;
        for (int e = 0; e < 64; ++e) s += sT[c * 64 + e] * Wkp[e * 64 + d];
        sG[i] = s * 0.125f;
    }
    __syncthreads();
    // softmax over d per row c; emit packed bf16 (att is B operand: K=c, N=d)
    if (tid < 64) {
        int c = tid;
        float mx = -3.402823e38f;
        for (int d = 0; d < 64; ++d) mx = fmaxf(mx, sG[c * 64 + d]);
        float sum = 0.f;
        for (int d = 0; d < 64; ++d) sum += expf(sG[c * 64 + d] - mx);
        float inv = 1.f / sum;
        __bf16* ap = att + ((size_t)(band * B_ + b)) * 4096;
        for (int d = 0; d < 64; ++d)
            ap[bpack_idx(c, d, 4)] = f2bf(expf(sG[c * 64 + d] - mx) * inv);
    }
}

// ---------------------------------------------------------------------------
// K7: v = mp_flat @ Wv  (131072 x 64 @ 64 x 64) -> bf16 row-major.
// ---------------------------------------------------------------------------
__global__ void k_vgemm(const __bf16* __restrict__ A, const __bf16* __restrict__ Wvp,
                        __bf16* __restrict__ V) {
    int wave = threadIdx.x >> 5, lane = threadIdx.x & 31;
    int row0 = (blockIdx.x * 8 + wave) * 16;
    v16bf a0 = load_a_rm(A + (size_t)row0 * C_, C_, lane);
    v16bf a1 = load_a_rm(A + (size_t)row0 * C_ + 32, C_, lane);
#pragma unroll
    for (int nt = 0; nt < 4; ++nt) {
        v8f acc = {};
        acc = wmma_bf16(a0, load_packed(Wvp + (size_t)(0 * 4 + nt) * 512, lane), acc);
        acc = wmma_bf16(a1, load_packed(Wvp + (size_t)(1 * 4 + nt) * 512, lane), acc);
        store_d_bf16(V + (size_t)row0 * C_ + nt * 16, C_, lane, acc);
    }
}

// ---------------------------------------------------------------------------
// K8: band_out[band] = v @ att[band][batch]  -> fp32 [band][131072][64]
// ---------------------------------------------------------------------------
__global__ void k_bandout(const __bf16* __restrict__ V, const __bf16* __restrict__ att,
                          float* __restrict__ bo) {
    int band = blockIdx.y;
    int wave = threadIdx.x >> 5, lane = threadIdx.x & 31;
    int row0 = (blockIdx.x * 8 + wave) * 16;
    int batch = row0 >> 14;                              // 16384 rows per batch
    const __bf16* Bp = att + ((size_t)(band * B_ + batch)) * 4096;  // packed
    const __bf16* Ap = V + (size_t)row0 * C_;
    float* outp = bo + ((size_t)band * NROWS_MP + row0) * C_;
    v16bf a0 = load_a_rm(Ap, C_, lane);
    v16bf a1 = load_a_rm(Ap + 32, C_, lane);
#pragma unroll
    for (int nt = 0; nt < 4; ++nt) {
        v8f acc = {};
        acc = wmma_bf16(a0, load_packed(Bp + (size_t)(0 * 4 + nt) * 512, lane), acc);
        acc = wmma_bf16(a1, load_packed(Bp + (size_t)(1 * 4 + nt) * 512, lane), acc);
        store_d_f32(outp + nt * 16, C_, lane, acc);
    }
}

// ---------------------------------------------------------------------------
// K9: IDWT along W: (LL',LH') -> lo2, (HL',HH') -> hi2; [B,128,256,C] fp32
// ---------------------------------------------------------------------------
__global__ void k_idwt_w(const float* __restrict__ bo, float* __restrict__ lo2,
                         float* __restrict__ hi2) {
    int pair = blockIdx.y;
    size_t BOSZ = (size_t)NROWS_MP * C_;
    const float* aB = bo + (size_t)(pair * 2 + 0) * BOSZ;   // approx coeffs
    const float* dB = bo + (size_t)(pair * 2 + 1) * BOSZ;   // detail coeffs
    int t = blockIdx.x * blockDim.x + threadIdx.x;
    int c = t & 63, wo = (t >> 6) & 255, h = (t >> 14) & 127, b = t >> 21;
    float acc = 0.f;
#pragma unroll
    for (int k = 0; k < 8; ++k) {
        int idx = (wo - k) & 255;
        if (!(idx & 1)) {
            int m = idx >> 1;
            size_t off = ((((size_t)b * HP_) + h) * WP_ + m) * C_ + c;
            acc += DEC_LO[k] * aB[off] + DEC_HI[k] * dB[off];
        }
    }
    float* dst = pair ? hi2 : lo2;
    dst[t] = acc;
}

// ---------------------------------------------------------------------------
// K10: IDWT along H + add path_a + LayerNorm(C) -> x_ln bf16 [B,256,256,C]
// ---------------------------------------------------------------------------
__global__ void k_idwt_h_ln(const float* __restrict__ pa, const float* __restrict__ lo2,
                            const float* __restrict__ hi2, const float* __restrict__ gamma,
                            const float* __restrict__ beta, __bf16* __restrict__ xln) {
    __shared__ float sv[256];
    int c = threadIdx.x & 63, g = threadIdx.x >> 6;
    int pix = blockIdx.x * 4 + g;
    int w = pix & 255, n = (pix >> 8) & 255, b = pix >> 16;
    float acc = pa[(size_t)pix * C_ + c];
#pragma unroll
    for (int k = 0; k < 8; ++k) {
        int idx = (n - k) & 255;
        if (!(idx & 1)) {
            int m = idx >> 1;
            size_t off = ((((size_t)b * HP_) + m) * W_ + w) * C_ + c;
            acc += DEC_LO[k] * lo2[off] + DEC_HI[k] * hi2[off];
        }
    }
    sv[threadIdx.x] = acc;
    __syncthreads();
    int g0 = g << 6;
    float sum = 0.f, ss = 0.f;
    for (int i = 0; i < 64; ++i) { float v = sv[g0 + i]; sum += v; ss += v * v; }
    float mu = sum * (1.f / 64.f);
    float var = ss * (1.f / 64.f) - mu * mu;
    float xn = (acc - mu) * rsqrtf(var + 1e-3f) * gamma[c] + beta[c];
    xln[(size_t)pix * C_ + c] = f2bf(xn);
}

// ---------------------------------------------------------------------------
// K11: fused MLP: out = gelu(x @ W1 + b1) @ W2 + b2.  Block 128 = 4 waves,
// 16 rows per wave; hidden staged bf16 through per-wave LDS (D -> A refragment).
// ---------------------------------------------------------------------------
__global__ void k_mlp(const __bf16* __restrict__ X, const __bf16* __restrict__ W1p,
                      const float* __restrict__ b1, const __bf16* __restrict__ W2p,
                      const float* __restrict__ b2, float* __restrict__ out) {
    __shared__ __align__(32) __bf16 sH[4 * 16 * 256];
    int wave = threadIdx.x >> 5, lane = threadIdx.x & 31;
    __bf16* sHw = sH + wave * 16 * 256;
    int row0 = (blockIdx.x * 4 + wave) * 16;
    int half = (lane >> 4) & 1, nn = lane & 15;

    v16bf a0 = load_a_rm(X + (size_t)row0 * C_, C_, lane);
    v16bf a1 = load_a_rm(X + (size_t)row0 * C_ + 32, C_, lane);

    // GEMM1: hidden = gelu(x @ W1 + b1); W1 packed tiles (kt*16 + jt)
#pragma unroll
    for (int jt = 0; jt < 16; ++jt) {
        v8f acc = {};
        acc = wmma_bf16(a0, load_packed(W1p + (size_t)(0 * 16 + jt) * 512, lane), acc);
        acc = wmma_bf16(a1, load_packed(W1p + (size_t)(1 * 16 + jt) * 512, lane), acc);
        int col = jt * 16 + nn;
        float bias = b1[col];
#pragma unroll
        for (int r = 0; r < 8; ++r) {
            float h = acc[r] + bias;
            float gel = 0.5f * h * (1.f + erff(h * 0.70710678118654752f));
            sHw[(r + 8 * half) * 256 + col] = f2bf(gel);
        }
    }

    // GEMM2: out = hidden @ W2 + b2 (K = 256); W2 packed tiles (ks*4 + nt)
    v8f oacc0 = {}, oacc1 = {}, oacc2 = {}, oacc3 = {};
#pragma unroll
    for (int ks = 0; ks < 8; ++ks) {
        v16bf af = load_a_rm(sHw + ks * 32, 256, lane);
        oacc0 = wmma_bf16(af, load_packed(W2p + (size_t)(ks * 4 + 0) * 512, lane), oacc0);
        oacc1 = wmma_bf16(af, load_packed(W2p + (size_t)(ks * 4 + 1) * 512, lane), oacc1);
        oacc2 = wmma_bf16(af, load_packed(W2p + (size_t)(ks * 4 + 2) * 512, lane), oacc2);
        oacc3 = wmma_bf16(af, load_packed(W2p + (size_t)(ks * 4 + 3) * 512, lane), oacc3);
    }
    float* outp = out + (size_t)row0 * C_;
    v8f oa[4] = {oacc0, oacc1, oacc2, oacc3};
#pragma unroll
    for (int nt = 0; nt < 4; ++nt) {
        float bias = b2[nt * 16 + nn];
#pragma unroll
        for (int r = 0; r < 8; ++r)
            outp[(r + 8 * half) * C_ + nt * 16 + nn] = oa[nt][r] + bias;
    }
}

// ---------------------------------------------------------------------------
// Host: workspace layout + launches
// ---------------------------------------------------------------------------
extern "C" void kernel_launch(void* const* d_in, const int* in_sizes, int n_in,
                              void* d_out, int out_size, void* d_ws, size_t ws_size,
                              hipStream_t stream) {
    const float* img   = (const float*)d_in[0];
    const float* ker   = (const float*)d_in[1];
    const float* Wv    = (const float*)d_in[2];
    const float* Wq    = (const float*)d_in[3];
    const float* Wk    = (const float*)d_in[4];
    const float* gamma = (const float*)d_in[5];
    const float* beta  = (const float*)d_in[6];
    const float* W1    = (const float*)d_in[7];
    const float* b1    = (const float*)d_in[8];
    const float* W2    = (const float*)d_in[9];
    const float* b2    = (const float*)d_in[10];
    float* out = (float*)d_out;

    char* p = (char*)d_ws;
    auto alloc = [&](size_t bytes) -> void* {
        void* r = (void*)p;
        p += (bytes + 255) & ~(size_t)255;
        return r;
    };
    float*  pa     = (float*)alloc((size_t)NROWS_FULL * C_ * 4);    // 134 MB
    float*  mp     = (float*)alloc((size_t)NROWS_MP * C_ * 4);      //  33 MB
    __bf16* mpb    = (__bf16*)alloc((size_t)NROWS_MP * C_ * 2);
    float*  lo     = (float*)alloc((size_t)B_ * 64 * WP_ * C_ * 4);
    float*  hi     = (float*)alloc((size_t)B_ * 64 * WP_ * C_ * 4);
    __bf16* bandsA = (__bf16*)alloc((size_t)4 * B_ * NB_ * C_ * 2); // A^T packed
    __bf16* bandsB = (__bf16*)alloc((size_t)4 * B_ * NB_ * C_ * 2); // B packed
    float*  G      = (float*)alloc((size_t)4 * B_ * 4096 * 4);
    __bf16* att    = (__bf16*)alloc((size_t)4 * B_ * 4096 * 2);     // B packed
    __bf16* vb     = (__bf16*)alloc((size_t)NROWS_MP * C_ * 2);
    float*  bo     = (float*)alloc((size_t)4 * NROWS_MP * C_ * 4);  // 134 MB
    float*  lo2    = (float*)alloc((size_t)B_ * HP_ * W_ * C_ * 4); //  67 MB
    float*  hi2    = (float*)alloc((size_t)B_ * HP_ * W_ * C_ * 4); //  67 MB
    __bf16* xln    = (__bf16*)alloc((size_t)NROWS_FULL * C_ * 2);   //  67 MB
    __bf16* wvp    = (__bf16*)alloc(4096 * 2);
    __bf16* w1p    = (__bf16*)alloc(16384 * 2);
    __bf16* w2p    = (__bf16*)alloc(16384 * 2);

    // weight conversion into B-packed layout
    k_cvt_bpack<<<16, 256, 0, stream>>>(Wv, wvp, 4096, 6);   // 64x64
    k_cvt_bpack<<<64, 256, 0, stream>>>(W1, w1p, 16384, 8);  // 64x256
    k_cvt_bpack<<<64, 256, 0, stream>>>(W2, w2p, 16384, 6);  // 256x64

    // path A + pooling + analysis DWT
    k_dwconv <<<(NROWS_FULL * C_) / 256, 256, 0, stream>>>(img, ker, pa);
    k_maxpool<<<(NROWS_MP * C_) / 256, 256, 0, stream>>>(img, mp, mpb);
    k_dwt_h  <<<(B_ * 64 * WP_ * C_) / 256, 256, 0, stream>>>(mp, lo, hi);
    k_dwt_w  <<<dim3((B_ * NB_ * C_) / 256, 2), 256, 0, stream>>>(lo, hi, bandsA, bandsB);

    // attention (WMMA Gram + softmax of Wq^T G Wk)
    k_gram<<<dim3(B_, 4), 256, 0, stream>>>(bandsA, bandsB, G);
    k_att <<<dim3(B_, 4), 256, 0, stream>>>(G, Wq, Wk, att);

    // value path (WMMA)
    k_vgemm  <<<NROWS_MP / 128, 256, 0, stream>>>(mpb, wvp, vb);
    k_bandout<<<dim3(NROWS_MP / 128, 4), 256, 0, stream>>>(vb, att, bo);

    // synthesis IDWT + add + LayerNorm
    k_idwt_w   <<<dim3((B_ * HP_ * W_ * C_) / 256, 2), 256, 0, stream>>>(bo, lo2, hi2);
    k_idwt_h_ln<<<NROWS_FULL / 4, 256, 0, stream>>>(pa, lo2, hi2, gamma, beta, xln);

    // fused MLP (WMMA), writes fp32 output
    k_mlp<<<NROWS_FULL / 64, 128, 0, stream>>>(xln, w1p, b1, w2p, b2, out);
}
